// MultiHeadAttention_31507880083696
// MI455X (gfx1250) — matrix operand order repaired and checked by reading, not compiled, not run on hardware
//
#include <hip/hip_runtime.h>
#include <hip/hip_bf16.h>

// ---------------------------------------------------------------------------
// MHA forward for MI455X (gfx1250, wave32, WMMA).
//   B=2, T=2048, C=1024, H=16, d_head=64
// Pipeline: qkv (f16 Q/K + transposed f16 V) -> flash attention -> out proj.
// All matmuls on v_wmma_f32_16x16x32_f16 (fp32 accumulate).
// Attention double-buffers K/V blocks into LDS with gfx1250 async loads
// (GLOBAL_LOAD_ASYNC_TO_LDS_B128 / ASYNCcnt).
// ---------------------------------------------------------------------------

#define D_MODEL   1024
#define NUM_HEADS 16
#define D_HEAD    64
#define BATCH     2
#define SEQ       2048
#define WSTRIDE   40   // halfs: weight-slab LDS column stride (80 B)
#define KSTRIDE   72   // halfs: K-block LDS row stride (144 B)
#define VSTRIDE   40   // halfs: V-block LDS row stride (80 B)

typedef __attribute__((ext_vector_type(16))) _Float16 v16h;
typedef __attribute__((ext_vector_type(8)))  _Float16 v8h;
typedef __attribute__((ext_vector_type(8)))  float    v8f;
typedef __attribute__((ext_vector_type(4)))  int      v4i;

union V16 { v16h v; v8h h[2]; };

static __device__ __forceinline__ v8f wmma_f16(v16h a, v16h b, v8f c) {
  return __builtin_amdgcn_wmma_f32_16x16x32_f16(false, a, false, b,
                                                (short)0, c, false, false);
}

// ---- gfx1250 async global->LDS copy (16 B per lane per op) ----------------
// builtin signature: (global v4i32*, LDS v4i32*, imm offset, imm cpol)
#if __has_builtin(__builtin_amdgcn_global_load_async_to_lds_b128)
#define ASYNC_KV 1
static __device__ __forceinline__ void async_cp16(const _Float16* g,
                                                  _Float16* l) {
  __builtin_amdgcn_global_load_async_to_lds_b128(
      (__attribute__((address_space(1))) v4i*)(void*)g,
      (__attribute__((address_space(3))) v4i*)(void*)l, 0, 0);
}
#else
#define ASYNC_KV 0
#endif

#if __has_builtin(__builtin_amdgcn_s_wait_asynccnt)
#define WAIT_ASYNC(n) __builtin_amdgcn_s_wait_asynccnt(n)
#else
#define WAIT_ASYNC(n) asm volatile("s_wait_asynccnt %0" ::"i"(n) : "memory")
#endif

// ---------------------------------------------------------------------------
// Kernel 1: Q/K/V projection.  grid.x = B*H*(SEQ/64), grid.y = kind (0=q,1=k,2=v)
// Block = 128 threads (4 waves); wave w owns rows t0 = chunk*64 + w*16.
// Weight K-slab [32 x 64] staged to LDS transposed ([n][k], f16) so each
// B fragment is a pair of ds_load_b128.
// ---------------------------------------------------------------------------
__global__ __launch_bounds__(128) void qkv_kernel(
    const float* __restrict__ x,
    const float* __restrict__ Wq, const float* __restrict__ Wk,
    const float* __restrict__ Wv,
    const float* __restrict__ bq, const float* __restrict__ bk,
    const float* __restrict__ bv,
    _Float16* __restrict__ Qo, _Float16* __restrict__ Ko,
    _Float16* __restrict__ Vt)
{
  __shared__ _Float16 lds_wt[D_HEAD * WSTRIDE];

  const int kind   = blockIdx.y;
  const int chunks = SEQ / 64;
  const int bh     = blockIdx.x / chunks;
  const int chunk  = blockIdx.x % chunks;
  const int b      = bh / NUM_HEADS;
  const int h      = bh % NUM_HEADS;

  const float* W    = (kind == 0) ? Wq : (kind == 1) ? Wk : Wv;
  const float* bias = (kind == 0) ? bq : (kind == 1) ? bk : bv;
  W    += (size_t)h * D_MODEL * D_HEAD;
  bias += h * D_HEAD;

  const int tid  = threadIdx.x;
  const int wave = tid >> 5;
  const int lane = tid & 31;
  const int row  = lane & 15;
  const int hi   = lane >> 4;
  const int t0   = chunk * 64 + wave * 16;

  const float* xrow = x + ((size_t)b * SEQ + (t0 + row)) * D_MODEL;

  v8f acc[4] = {};

  for (int c0 = 0; c0 < D_MODEL; c0 += 32) {
    // --- stage W[c0..c0+32)[0..64) transposed -> lds_wt[n][k] --------------
    #pragma unroll
    for (int t = tid; t < 256; t += 128) {
      const int n  = t & 63;
      const int kc = t >> 6;
      const float* wp = W + (size_t)(c0 + kc * 8) * D_HEAD + n;
      v8h tmp;
      #pragma unroll
      for (int i = 0; i < 8; ++i) tmp[i] = (_Float16)wp[(size_t)i * D_HEAD];
      *(v8h*)&lds_wt[n * WSTRIDE + kc * 8] = tmp;
    }
    __syncthreads();

    // --- A fragment (16x32 of x), per ISA 16-bit A layout ------------------
    const int kb = hi * 8;
    const float4 fa = *(const float4*)(xrow + c0 + kb);
    const float4 fb = *(const float4*)(xrow + c0 + kb + 4);
    const float4 fc = *(const float4*)(xrow + c0 + kb + 16);
    const float4 fd = *(const float4*)(xrow + c0 + kb + 20);
    v16h a;
    a[0]  = (_Float16)fa.x; a[1]  = (_Float16)fa.y;
    a[2]  = (_Float16)fa.z; a[3]  = (_Float16)fa.w;
    a[4]  = (_Float16)fb.x; a[5]  = (_Float16)fb.y;
    a[6]  = (_Float16)fb.z; a[7]  = (_Float16)fb.w;
    a[8]  = (_Float16)fc.x; a[9]  = (_Float16)fc.y;
    a[10] = (_Float16)fc.z; a[11] = (_Float16)fc.w;
    a[12] = (_Float16)fd.x; a[13] = (_Float16)fd.y;
    a[14] = (_Float16)fd.z; a[15] = (_Float16)fd.w;

    // --- load ALL 4 B fragments first (one dscnt wait), then 4 WMMAs -------
    const int kw = hi * 8;
    V16 bf[4];
    #pragma unroll
    for (int j = 0; j < 4; ++j) {
      const _Float16* bp = &lds_wt[(j * 16 + row) * WSTRIDE + kw];
      bf[j].h[0] = *(const v8h*)bp;
      bf[j].h[1] = *(const v8h*)(bp + 16);
    }
    #pragma unroll
    for (int j = 0; j < 4; ++j)
      acc[j] = wmma_f16(a, bf[j].v, acc[j]);
    __syncthreads();
  }

  // --- epilogue: bias, (Q-scale), store f16 --------------------------------
  const size_t basebh = (size_t)bh * SEQ * D_HEAD;
  #pragma unroll
  for (int j = 0; j < 4; ++j) {
    const int n    = j * 16 + row;
    const float bb = bias[n];
    #pragma unroll
    for (int r = 0; r < 8; ++r) {
      const int t    = t0 + r + hi * 8;
      const float vv = acc[j][r] + bb;
      if (kind == 0)
        Qo[basebh + (size_t)t * D_HEAD + n] = (_Float16)(vv * 0.03125f); // 1/sqrt(1024)
      else if (kind == 1)
        Ko[basebh + (size_t)t * D_HEAD + n] = (_Float16)vv;
      else
        Vt[basebh + (size_t)n * SEQ + t] = (_Float16)vv;                 // transposed
    }
  }
}

// ---------------------------------------------------------------------------
// Kernel 2: flash attention. One wave32 per (b,h,t-tile-of-16).
// Streams s in blocks of 32: K/V blocks double-buffered in LDS via async
// global->LDS b128 copies; 4 WMMAs for S = Q K^T, online softmax in
// registers, P through LDS (C-layout -> A-layout), 4 WMMAs for P @ V.
// Single-wave workgroup: barriers are free (S_NOP per ISA).
// ---------------------------------------------------------------------------
__global__ __launch_bounds__(32) void attn_kernel(
    const _Float16* __restrict__ Q, const _Float16* __restrict__ K,
    const _Float16* __restrict__ Vt, _Float16* __restrict__ AO)
{
  __shared__ _Float16 lds_p[16 * 32];
#if ASYNC_KV
  __shared__ _Float16 kbuf[2][32 * KSTRIDE];
  __shared__ _Float16 vbuf[2][64 * VSTRIDE];
#endif

  const int tiles = SEQ / 16;
  const int bh    = blockIdx.x / tiles;
  const int t0    = (blockIdx.x % tiles) * 16;
  const int b     = bh / NUM_HEADS;
  const int h     = bh % NUM_HEADS;

  const _Float16* Qh = Q  + (size_t)bh * SEQ * D_HEAD;
  const _Float16* Kh = K  + (size_t)bh * SEQ * D_HEAD;
  const _Float16* Vh = Vt + (size_t)bh * D_HEAD * SEQ;

  const int lane = threadIdx.x & 31;
  const int row  = lane & 15;
  const int hi   = lane >> 4;
  const int kb   = hi * 8;

  // Q fragments for K-dim d = 0..31 and 32..63 (held for the whole loop)
  V16 qa0, qa1;
  {
    const _Float16* qrow = Qh + (size_t)(t0 + row) * D_HEAD;
    qa0.h[0] = *(const v8h*)(qrow + hi * 16);
    qa0.h[1] = *(const v8h*)(qrow + hi * 16 + 8);
    qa1.h[0] = *(const v8h*)(qrow + 32 + hi * 16);
    qa1.h[1] = *(const v8h*)(qrow + 32 + hi * 16 + 8);
  }

  float m[8], l[8];
  #pragma unroll
  for (int r = 0; r < 8; ++r) { m[r] = -__builtin_inff(); l[r] = 0.0f; }
  v8f acc[4] = {};

#if ASYNC_KV
  // issue the first K/V block (16 async b128 ops = 8 KB of f16)
  {
    const _Float16* g = Kh + (size_t)lane * D_HEAD;       // row s = lane
    _Float16* lk = &kbuf[0][lane * KSTRIDE];
    #pragma unroll
    for (int i = 0; i < 8; ++i) async_cp16(g + i * 8, lk + i * 8);
    #pragma unroll
    for (int rr = 0; rr < 2; ++rr) {
      const int d = lane + rr * 32;
      const _Float16* gv = Vh + (size_t)d * SEQ;          // cols s = 0..31
      _Float16* lv = &vbuf[0][d * VSTRIDE];
      #pragma unroll
      for (int i = 0; i < 4; ++i) async_cp16(gv + i * 8, lv + i * 8);
    }
  }
  int buf = 0;
#endif

  for (int s0 = 0; s0 < SEQ; s0 += 32) {
#if ASYNC_KV
    // ---- issue next block into the other buffer, retire current ----------
    if (s0 + 32 < SEQ) {
      const int nb = buf ^ 1;
      const _Float16* g = Kh + (size_t)(s0 + 32 + lane) * D_HEAD;
      _Float16* lk = &kbuf[nb][lane * KSTRIDE];
      #pragma unroll
      for (int i = 0; i < 8; ++i) async_cp16(g + i * 8, lk + i * 8);
      #pragma unroll
      for (int rr = 0; rr < 2; ++rr) {
        const int d = lane + rr * 32;
        const _Float16* gv = Vh + (size_t)d * SEQ + s0 + 32;
        _Float16* lv = &vbuf[nb][d * VSTRIDE];
        #pragma unroll
        for (int i = 0; i < 4; ++i) async_cp16(gv + i * 8, lv + i * 8);
      }
      WAIT_ASYNC(16);   // 16 newest (next block) may remain in flight
    } else {
      WAIT_ASYNC(0);
    }
    const _Float16* kB = kbuf[buf];
    const _Float16* vB = vbuf[buf];
#endif

    // ---- scores: S[16 x 32] via 4 WMMAs (two 16-col s-tiles) --------------
    v8f S0 = {}, S1 = {};
    {
#if ASYNC_KV
      V16 kf0, kf1, kg0, kg1;
      const _Float16* kp0 = kB + row * KSTRIDE + hi * 16;
      kf0.h[0] = *(const v8h*)kp0;        kf0.h[1] = *(const v8h*)(kp0 + 8);
      kf1.h[0] = *(const v8h*)(kp0 + 32); kf1.h[1] = *(const v8h*)(kp0 + 40);
      const _Float16* kp1 = kB + (16 + row) * KSTRIDE + hi * 16;
      kg0.h[0] = *(const v8h*)kp1;        kg0.h[1] = *(const v8h*)(kp1 + 8);
      kg1.h[0] = *(const v8h*)(kp1 + 32); kg1.h[1] = *(const v8h*)(kp1 + 40);
      S0 = wmma_f16(qa0.v, kf0.v, S0);
      S0 = wmma_f16(qa1.v, kf1.v, S0);
      S1 = wmma_f16(qa0.v, kg0.v, S1);
      S1 = wmma_f16(qa1.v, kg1.v, S1);
#else
      const _Float16* kp0 = Kh + (size_t)(s0 + row) * D_HEAD + hi * 16;
      v16h kf0 = *(const v16h*)kp0;
      v16h kf1 = *(const v16h*)(kp0 + 32);
      S0 = wmma_f16(qa0.v, kf0, S0);
      S0 = wmma_f16(qa1.v, kf1, S0);
      const _Float16* kp1 = Kh + (size_t)(s0 + 16 + row) * D_HEAD + hi * 16;
      v16h kg0 = *(const v16h*)kp1;
      v16h kg1 = *(const v16h*)(kp1 + 32);
      S1 = wmma_f16(qa0.v, kg0, S1);
      S1 = wmma_f16(qa1.v, kg1, S1);
#endif
    }

    // ---- online softmax update (per row r + 8*hi) -------------------------
    float alpha[8];
    #pragma unroll
    for (int r = 0; r < 8; ++r) {
      float v0 = S0[r], v1 = S1[r];
      float bm = fmaxf(v0, v1);
      #pragma unroll
      for (int mask = 1; mask < 16; mask <<= 1)
        bm = fmaxf(bm, __shfl_xor(bm, mask, 32));
      const float mn = fmaxf(m[r], bm);
      const float a_ = __expf(m[r] - mn);
      const float p0 = __expf(v0 - mn);
      const float p1 = __expf(v1 - mn);
      float rs = p0 + p1;
      #pragma unroll
      for (int mask = 1; mask < 16; mask <<= 1)
        rs += __shfl_xor(rs, mask, 32);
      l[r] = l[r] * a_ + rs;
      m[r] = mn;
      alpha[r] = a_;
      lds_p[(r + hi * 8) * 32 + row]      = (_Float16)p0;
      lds_p[(r + hi * 8) * 32 + 16 + row] = (_Float16)p1;
    }
    #pragma unroll
    for (int j = 0; j < 4; ++j)
      #pragma unroll
      for (int r = 0; r < 8; ++r) acc[j][r] *= alpha[r];

    __syncthreads();  // single wave -> S_NOP; LDS store->load ordering

    // ---- P (A-layout) from LDS as 2x b128, acc += P @ V -------------------
    V16 pa;
    pa.h[0] = *(const v8h*)&lds_p[row * 32 + hi * 16];
    pa.h[1] = *(const v8h*)&lds_p[row * 32 + hi * 16 + 8];
    #pragma unroll
    for (int j = 0; j < 4; ++j) {
#if ASYNC_KV
      V16 vb;
      const _Float16* vp = vB + (j * 16 + row) * VSTRIDE + hi * 16;
      vb.h[0] = *(const v8h*)vp;
      vb.h[1] = *(const v8h*)(vp + 8);
      acc[j] = wmma_f16(pa.v, vb.v, acc[j]);
#else
      const _Float16* vp = Vh + (size_t)(j * 16 + row) * SEQ + s0 + hi * 16;
      v16h vb = *(const v16h*)vp;
      acc[j] = wmma_f16(pa.v, vb, acc[j]);
#endif
    }
    __syncthreads();
#if ASYNC_KV
    buf ^= 1;
#endif
  }

  // ---- normalize, store concat-head layout [B,T,H*64] as f16 --------------
  float rl[8];
  #pragma unroll
  for (int r = 0; r < 8; ++r) rl[r] = 1.0f / l[r];
  #pragma unroll
  for (int j = 0; j < 4; ++j) {
    const int d = j * 16 + row;
    #pragma unroll
    for (int r = 0; r < 8; ++r) {
      const int t = t0 + r + hi * 8;
      AO[((size_t)b * SEQ + t) * (NUM_HEADS * D_HEAD) + h * D_HEAD + d] =
          (_Float16)(acc[j][r] * rl[r]);
    }
  }
}

// ---------------------------------------------------------------------------
// Kernel 3: output projection [B*T,1024] @ Wo[1024,1024] + bo -> f32 out.
// grid.x = M/64, grid.y = N/64. 4 waves/block; Wo slab staged transposed.
// ---------------------------------------------------------------------------
__global__ __launch_bounds__(128) void oproj_kernel(
    const _Float16* __restrict__ AO, const float* __restrict__ Wo,
    const float* __restrict__ bo, float* __restrict__ out)
{
  __shared__ _Float16 lds_wt[64 * WSTRIDE];

  const int n0b  = blockIdx.y * 64;
  const int tid  = threadIdx.x;
  const int wave = tid >> 5;
  const int lane = tid & 31;
  const int row  = lane & 15;
  const int hi   = lane >> 4;
  const int m0   = blockIdx.x * 64 + wave * 16;

  const _Float16* arow = AO + (size_t)(m0 + row) * D_MODEL;

  v8f acc[4] = {};

  for (int c0 = 0; c0 < D_MODEL; c0 += 32) {
    // stage Wo[c0..c0+32)[n0b..n0b+64) transposed -> lds_wt[n][k]
    #pragma unroll
    for (int t = tid; t < 256; t += 128) {
      const int n  = t & 63;
      const int kc = t >> 6;
      const float* wp = Wo + (size_t)(c0 + kc * 8) * D_MODEL + n0b + n;
      v8h tmp;
      #pragma unroll
      for (int i = 0; i < 8; ++i) tmp[i] = (_Float16)wp[(size_t)i * D_MODEL];
      *(v8h*)&lds_wt[n * WSTRIDE + kc * 8] = tmp;
    }
    __syncthreads();

    const int kb = hi * 8;
    V16 a;
    a.h[0] = *(const v8h*)(arow + c0 + kb);
    a.h[1] = *(const v8h*)(arow + c0 + kb + 16);

    const int kw = hi * 8;
    V16 bf[4];
    #pragma unroll
    for (int j = 0; j < 4; ++j) {
      const _Float16* bp = &lds_wt[(j * 16 + row) * WSTRIDE + kw];
      bf[j].h[0] = *(const v8h*)bp;
      bf[j].h[1] = *(const v8h*)(bp + 16);
    }
    #pragma unroll
    for (int j = 0; j < 4; ++j)
      acc[j] = wmma_f16(a.v, bf[j].v, acc[j]);
    __syncthreads();
  }

  #pragma unroll
  for (int j = 0; j < 4; ++j) {
    const int n    = n0b + j * 16 + row;
    const float bb = bo[n];
    #pragma unroll
    for (int r = 0; r < 8; ++r)
      out[(size_t)(m0 + r + hi * 8) * D_MODEL + n] = acc[j][r] + bb;
  }
}

// ---------------------------------------------------------------------------
extern "C" void kernel_launch(void* const* d_in, const int* in_sizes, int n_in,
                              void* d_out, int out_size, void* d_ws,
                              size_t ws_size, hipStream_t stream)
{
  const float* x  = (const float*)d_in[0];
  const float* Wq = (const float*)d_in[1];
  const float* Wk = (const float*)d_in[2];
  const float* Wv = (const float*)d_in[3];
  const float* bq = (const float*)d_in[4];
  const float* bk = (const float*)d_in[5];
  const float* bv = (const float*)d_in[6];
  const float* Wo = (const float*)d_in[7];
  const float* bo = (const float*)d_in[8];
  float* out = (float*)d_out;

  // workspace: 4 f16 buffers of B*H*T*64 = 4,194,304 halfs each (32 MB total)
  const size_t SEG = (size_t)BATCH * NUM_HEADS * SEQ * D_HEAD;
  _Float16* ws = (_Float16*)d_ws;
  _Float16* Q  = ws;
  _Float16* K  = ws + SEG;
  _Float16* Vt = ws + 2 * SEG;
  _Float16* AO = ws + 3 * SEG;

  dim3 g1(BATCH * NUM_HEADS * (SEQ / 64), 3);
  qkv_kernel<<<g1, 128, 0, stream>>>(x, Wq, Wk, Wv, bq, bk, bv, Q, K, Vt);

  attn_kernel<<<BATCH * NUM_HEADS * (SEQ / 16), 32, 0, stream>>>(Q, K, Vt, AO);

  dim3 g3((BATCH * SEQ) / 64, D_MODEL / 64);
  oproj_kernel<<<g3, 128, 0, stream>>>(AO, Wo, bo, out);
}
